// MessagePassing_5884105196042
// MI455X (gfx1250) — matrix operand order repaired and checked by reading, not compiled, not run on hardware
//
#include <hip/hip_runtime.h>

#define NN 512
#define BB 2
#define EF 64
#define GF 192                      // 3*EF
#define ROWS (BB*NN*NN)             // 524288
#define TILES (ROWS/16)             // 32768
#define WPACK_DW (48*32*8)          // 12288 dwords = 48KB (24 col-tiles x 2 K-halves)

typedef __attribute__((ext_vector_type(16))) __bf16       v16bf;
typedef __attribute__((ext_vector_type(8)))  float        v8f;
typedef __attribute__((ext_vector_type(4)))  float        f32x4;
typedef __attribute__((ext_vector_type(4)))  unsigned int u32x4;

// workspace layout (units of float)
#define WS_WC    0                          // 192*64
#define WS_WPACK (WS_WC + GF*EF)            // 12288 dwords
#define WS_T1    (WS_WPACK + WPACK_DW)      // 1024*64
#define WS_T2    (WS_T1 + BB*NN*EF)
#define WS_U1    (WS_T2 + BB*NN*EF)         // 1024*192
#define WS_U2P   (WS_U1 + BB*NN*GF)

__device__ __forceinline__ unsigned short f2bf(float x) {
    union { float f; unsigned u; } c; c.f = x;
    unsigned r = c.u + 0x7FFFu + ((c.u >> 16) & 1u);   // RNE
    return (unsigned short)(r >> 16);
}

// fast transcendentals: v_exp_f32 is exp2 on AMD; v_rcp_f32 ~1ulp
#define LOG2E 1.4426950408889634f
__device__ __forceinline__ float sigmoidf_(float x) {
    return __builtin_amdgcn_rcpf(1.f + __builtin_amdgcn_exp2f(-LOG2E * x));
}
__device__ __forceinline__ float tanhf_(float x) {
#if defined(__gfx1250__) && __has_builtin(__builtin_amdgcn_tanhf)
    return __builtin_amdgcn_tanhf(x);                       // v_tanh_f32
#else
    return 2.f * __builtin_amdgcn_rcpf(1.f + __builtin_amdgcn_exp2f(-2.f * LOG2E * x)) - 1.f;
#endif
}

// ---- prep 1: t1[bn,e]=h[bn,:]·W1[e,:],  t2[bn,e]=h[bn,:]·W2[e,:] ----
__global__ void k_t12(const float* __restrict__ h, const float* __restrict__ W_lin,
                      float* __restrict__ t1, float* __restrict__ t2) {
    int idx = blockIdx.x * blockDim.x + threadIdx.x;
    if (idx >= BB*NN*EF) return;
    int e = idx & 63, bn = idx >> 6;
    const float* hr = h + bn * EF;
    float s1 = 0.f, s2 = 0.f;
    for (int f = 0; f < 64; ++f) {
        float hv = hr[f];
        s1 += hv * W_lin[e*192 + f];
        s2 += hv * W_lin[e*192 + 128 + f];
    }
    t1[idx] = s1; t2[idx] = s2;
}

// ---- prep 2: Wc = w_ih (192x64) @ We (64x64), We[e,k]=W_lin[e,64+k] ----
__global__ void k_wc(const float* __restrict__ w_ih, const float* __restrict__ W_lin,
                     float* __restrict__ Wc) {
    int idx = blockIdx.x * blockDim.x + threadIdx.x;
    if (idx >= GF*EF) return;
    int k = idx & 63, g = idx >> 6;
    float s = 0.f;
    for (int e = 0; e < 64; ++e) s += w_ih[g*64 + e] * W_lin[e*192 + 64 + k];
    Wc[idx] = s;
}

// ---- prep 3: u1[bn,g]=w_ih·t1[bn],  u2p[bn,g]=w_ih·t2[bn] + w_ih·b_lin ----
__global__ void k_u(const float* __restrict__ w_ih, const float* __restrict__ b_lin,
                    const float* __restrict__ t1, const float* __restrict__ t2,
                    float* __restrict__ u1, float* __restrict__ u2p) {
    int idx = blockIdx.x * blockDim.x + threadIdx.x;
    if (idx >= BB*NN*GF) return;
    int g = idx % GF, bn = idx / GF;
    float s1 = 0.f, s2 = 0.f, sb = 0.f;
    for (int e = 0; e < 64; ++e) {
        float w = w_ih[g*64 + e];
        s1 += w * t1[bn*64 + e];
        s2 += w * t2[bn*64 + e];
        sb += w * b_lin[e];
    }
    u1[idx] = s1; u2p[idx] = s2 + sb;
}

// ---- prep 4: shuffle W_big=[Wc; w_hh] (384x64) into WMMA-B register layout ----
// B (32x16) 16-bit layout: lane<16 -> col=lane,   K = 0..7 (v0-3), 16..23 (v4-7)
//                          lane>=16 -> col=lane-16, K = 8..15 (v0-3), 24..31 (v4-7)
// Wpack dword index = ((nt*2+kh)*32 + lane)*8 + v
__global__ void k_pack(const float* __restrict__ Wc, const float* __restrict__ w_hh,
                       unsigned int* __restrict__ Wpack) {
    int idx = blockIdx.x * blockDim.x + threadIdx.x;
    if (idx >= WPACK_DW) return;
    int v = idx & 7, lane = (idx >> 3) & 31, t = idx >> 8;
    int nt = t >> 1, kh = t & 1;
    int g  = nt*16 + (lane & 15);
    int kb = kh*32 + ((lane < 16) ? 0 : 8) + 2*v + ((v >= 4) ? 8 : 0);
    float w0 = (g < GF) ? Wc[g*64 + kb]     : w_hh[(g-GF)*64 + kb];
    float w1 = (g < GF) ? Wc[g*64 + kb + 1] : w_hh[(g-GF)*64 + kb + 1];
    Wpack[idx] = (unsigned)f2bf(w0) | ((unsigned)f2bf(w1) << 16);
}

// ---- main: one wave per 16-row tile; 16x384 bf16 GEMM + fused GRU/SiLU ----
__global__ void __launch_bounds__(256) k_main(
        const float* __restrict__ edge, const float* __restrict__ adj,
        const float* __restrict__ b_ih, const float* __restrict__ b_hh,
        const unsigned int* __restrict__ Wpack,
        const float* __restrict__ u1, const float* __restrict__ u2p,
        float* __restrict__ out) {
    __shared__ __align__(16) unsigned int sW[WPACK_DW];
    for (int t = threadIdx.x; t < WPACK_DW; t += 256) sW[t] = Wpack[t];
    __syncthreads();

    const int  l    = threadIdx.x & 31;
    const int  wid  = threadIdx.x >> 5;
    const long tile = (long)blockIdx.x * 8 + wid;
    const long rbase = tile * 16;
    const int  b     = (int)(rbase >> 18);          // N*N = 2^18
    const int  rem   = (int)(rbase & 262143);
    const int  i     = rem >> 9;
    const int  jbase = rem & 511;

    // ---- A operand: 16x64 edge tile in WMMA-A lane layout, fp32 -> bf16 ----
    // A (16x32) 16-bit layout: lane holds row M=lane%16;
    //   lane<16:  K = 0..7 (v0-3), 16..23 (v4-7);  lane>=16: K = 8..15, 24..31
    const float* erow = edge + (rbase + (l & 15)) * 64;
    const int kb0 = (l < 16) ? 0 : 8;
    v16bf aV[2];
#pragma unroll
    for (int kh = 0; kh < 2; ++kh) {
        int base = kh*32 + kb0;
        f32x4 q0 = *(const f32x4*)(erow + base);
        f32x4 q1 = *(const f32x4*)(erow + base + 4);
        f32x4 q2 = *(const f32x4*)(erow + base + 16);
        f32x4 q3 = *(const f32x4*)(erow + base + 20);
        v16bf a;
#pragma unroll
        for (int q = 0; q < 4; ++q) {
            a[q]      = (__bf16)q0[q];
            a[4 + q]  = (__bf16)q1[q];
            a[8 + q]  = (__bf16)q2[q];
            a[12 + q] = (__bf16)q3[q];
        }
        aV[kh] = a;
    }

    const int voff = (l < 16) ? 0 : 8;
    f32x4 adjA = *(const f32x4*)(adj + rbase + voff);
    f32x4 adjB = *(const f32x4*)(adj + rbase + voff + 4);

#pragma unroll
    for (int ft = 0; ft < 4; ++ft) {
        // 6 output 16x16 tiles needed for this feature group:
        // gi_r=ft, gi_z=ft+4, gi_n=ft+8, gh_r=ft+12, gh_z=ft+16, gh_n=ft+20
        v8f acc[6];
#pragma unroll
        for (int s = 0; s < 6; ++s) {
            int nt = ft + s*4;
            v8f c;
#pragma unroll
            for (int q = 0; q < 8; ++q) c[q] = 0.f;
#pragma unroll
            for (int kh = 0; kh < 2; ++kh) {
                const unsigned int* bp = &sW[((nt*2 + kh)*32 + l) * 8];
                union { u32x4 q[2]; v16bf v; } bu;
                bu.q[0] = *(const u32x4*)bp;
                bu.q[1] = *(const u32x4*)(bp + 4);
                c = __builtin_amdgcn_wmma_f32_16x16x32_bf16(
                        false, aV[kh], false, bu.v, (short)0, c, false, false);
            }
            acc[s] = c;
        }

        const int f = ft*16 + (l & 15);
        const float* u2b = u2p + (size_t)(b*NN + i) * GF;
        float u2r = u2b[f], u2z = u2b[64 + f], u2n = u2b[128 + f];
        float bir = b_ih[f], biz = b_ih[64 + f], bin_ = b_ih[128 + f];
        float bhr = b_hh[f], bhz = b_hh[64 + f], bhn  = b_hh[128 + f];

#pragma unroll
        for (int v = 0; v < 8; ++v) {
            int   m  = voff + v;
            float av = (v < 4) ? adjA[v] : adjB[v - 4];
            const float* u1r = u1 + (size_t)(b*NN + jbase + m) * GF;
            float ir  = av * (acc[0][v] + u1r[f]       + u2r) + bir;
            float iz  = av * (acc[1][v] + u1r[64 + f]  + u2z) + biz;
            float inn = av * (acc[2][v] + u1r[128 + f] + u2n) + bin_;
            float hr  = acc[3][v] + bhr;
            float hz  = acc[4][v] + bhz;
            float hn  = acc[5][v] + bhn;
            float r  = sigmoidf_(ir + hr);
            float z  = sigmoidf_(iz + hz);
            float n  = tanhf_(inn + r * hn);
            float ev = edge[(rbase + m) * 64 + f];
            float nh = (1.f - z) * n + z * ev;
            out[(rbase + m) * 64 + f] = nh * sigmoidf_(nh);
        }
    }
}

extern "C" void kernel_launch(void* const* d_in, const int* in_sizes, int n_in,
                              void* d_out, int out_size, void* d_ws, size_t ws_size,
                              hipStream_t stream) {
    (void)in_sizes; (void)n_in; (void)out_size; (void)ws_size;
    const float* h     = (const float*)d_in[0];
    const float* edge  = (const float*)d_in[1];
    const float* adj   = (const float*)d_in[2];
    const float* W_lin = (const float*)d_in[3];
    const float* b_lin = (const float*)d_in[4];
    const float* w_ih  = (const float*)d_in[5];
    const float* w_hh  = (const float*)d_in[6];
    const float* b_ih  = (const float*)d_in[7];
    const float* b_hh  = (const float*)d_in[8];
    float* out = (float*)d_out;
    float* ws  = (float*)d_ws;

    float*        Wc    = ws + WS_WC;
    unsigned int* Wpack = (unsigned int*)(ws + WS_WPACK);
    float*        t1    = ws + WS_T1;
    float*        t2    = ws + WS_T2;
    float*        u1    = ws + WS_U1;
    float*        u2p   = ws + WS_U2P;

    k_t12 <<<(BB*NN*EF + 255)/256, 256, 0, stream>>>(h, W_lin, t1, t2);
    k_wc  <<<(GF*EF   + 255)/256, 256, 0, stream>>>(w_ih, W_lin, Wc);
    k_u   <<<(BB*NN*GF + 255)/256, 256, 0, stream>>>(w_ih, b_lin, t1, t2, u1, u2p);
    k_pack<<<(WPACK_DW + 255)/256, 256, 0, stream>>>(Wc, w_hh, Wpack);
    k_main<<<TILES/8, 256, 0, stream>>>(edge, adj, b_ih, b_hh, Wpack, u1, u2p, out);
}
